// QRNNAtt_1022202216448
// MI455X (gfx1250) — matrix operand order repaired and checked by reading, not compiled, not run on hardware
//
#include <hip/hip_runtime.h>

// ---------------------------------------------------------------------------
// QRNN + attention pooling for MI455X (gfx1250, wave32, WMMA).
// GEMM path: bf16 WMMA (v_wmma_f32_16x16x32_bf16), f32 accumulate, 32-row
// M tiles so each B fragment (L2-resident conv_w) feeds two WMMAs.
// Recurrence: 3-pass chunked linear scan; attention logits fused into pass 3
// with a single barrier per block.
// ---------------------------------------------------------------------------

typedef __attribute__((ext_vector_type(16))) __bf16 v16bf;
typedef __attribute__((ext_vector_type(8)))  __bf16 v8bf;
typedef __attribute__((ext_vector_type(8)))  float  v8f;

namespace qrnn {
constexpr int B  = 32, S = 4096, E = 256, H = 256;
constexpr int N2 = 2 * H;            // 512 cols: z (0..255), f (256..511); o is dead
constexpr int M  = B * S;            // 131072 token rows
constexpr int MROWS = 32;            // rows per block (2 x 16-row WMMA tiles)
constexpr int MB = M / MROWS;        // 4096 blocks
constexpr int CHUNKS = 64, CL = S / CHUNKS;   // 64 chunks x 64 steps
constexpr int CONVW_ROWS = 3 * H;    // 768 (rows 0..511 consumed)
}

__device__ __forceinline__ float waveReduceSum(float v) {
#pragma unroll
  for (int off = 16; off > 0; off >>= 1) v += __shfl_xor(v, off, 32);
  return v;
}

// --------------------------- conv_w f32 -> bf16 -----------------------------
__global__ void convw_to_bf16(const float* __restrict__ w, __bf16* __restrict__ o, int n) {
  int i = blockIdx.x * blockDim.x + threadIdx.x;
  if (i < n) o[i] = (__bf16)w[i];
}

// ------------- fused embedding-gather + GEMM (E -> 2H) + activations --------
// Block: 256 threads = 8 waves; owns a 32-row M tile (two 16-row WMMA tiles).
// Wave w owns N tiles [4w, 4w+4). K = 256 -> 8 WMMA steps of K=32; each B
// fragment is reused by both M tiles (2 WMMAs per 1 KB of L2 B traffic).
__global__ __launch_bounds__(256)
void gemm_zf(const int* __restrict__ X, const float* __restrict__ emb,
             const __bf16* __restrict__ wbf, const float* __restrict__ convb,
             float* __restrict__ zf) {
  using namespace qrnn;
  __shared__ __align__(16) __bf16 lA[MROWS][E + 8];   // 32 gathered rows, bf16

  const int mt = blockIdx.x;

  // Stage A tile: gather 32 embedding rows, convert f32 -> bf16 into LDS.
  {
    const int r = threadIdx.x >> 3;          // row 0..31
    const int c = (threadIdx.x & 7) * 32;    // 32 columns per thread
    const int tok = X[mt * MROWS + r];
    const float4* src = (const float4*)(emb + (long long)tok * E + c);
#pragma unroll
    for (int q = 0; q < 8; ++q) {
      float4 v = src[q];
      lA[r][c + q * 4 + 0] = (__bf16)v.x;
      lA[r][c + q * 4 + 1] = (__bf16)v.y;
      lA[r][c + q * 4 + 2] = (__bf16)v.z;
      lA[r][c + q * 4 + 3] = (__bf16)v.w;
    }
  }
  __syncthreads();

  const int lane = threadIdx.x & 31;
  const int wv   = threadIdx.x >> 5;   // wave 0..7
  const int half = lane >> 4;          // lane group 0/1
  const int ln16 = lane & 15;

  // Accumulators: [m-tile][n-tile], initialized with conv bias (bcast along M).
  v8f acc[2][4];
#pragma unroll
  for (int nt = 0; nt < 4; ++nt) {
    const int ng = (wv * 4 + nt) * 16 + ln16;   // output column 0..511
    const float bias = convb[ng];
#pragma unroll
    for (int i = 0; i < 8; ++i) { acc[0][nt][i] = bias; acc[1][nt][i] = bias; }
  }

#pragma unroll
  for (int kk = 0; kk < E / 32; ++kk) {
    // A fragments (16-bit A 16x32 layout, ISA 7.12.2): lane -> row ln16,
    // elems 0..7 = K[kk*32 + half*8 + j], elems 8..15 = +16.
    v16bf afrag[2];
#pragma unroll
    for (int p = 0; p < 2; ++p) {
      const __bf16* rp = &lA[p * 16 + ln16][kk * 32 + half * 8];
      const v8bf lo = *(const v8bf*)(rp);
      const v8bf hi = *(const v8bf*)(rp + 16);
#pragma unroll
      for (int i = 0; i < 8; ++i) { afrag[p][i] = lo[i]; afrag[p][i + 8] = hi[i]; }
    }
#pragma unroll
    for (int nt = 0; nt < 4; ++nt) {
      // B fragment (32x16, row K striped across lanes within VGPR):
      // lane -> column ng; elems 0..7 = K[kk*32 + half*8 + j], elems 8..15 = +16
      // in row-major conv_w -> two 16B global loads per lane (L2-resident).
      const int ng = (wv * 4 + nt) * 16 + ln16;
      const __bf16* bp = wbf + (long long)ng * E + kk * 32 + half * 8;
      const v8bf blo = *(const v8bf*)(bp);
      const v8bf bhi = *(const v8bf*)(bp + 16);
      v16bf bfrag;
#pragma unroll
      for (int i = 0; i < 8; ++i) { bfrag[i] = blo[i]; bfrag[i + 8] = bhi[i]; }
      acc[0][nt] = __builtin_amdgcn_wmma_f32_16x16x32_bf16(
          false, afrag[0], false, bfrag, (short)0, acc[0][nt], false, false);
      acc[1][nt] = __builtin_amdgcn_wmma_f32_16x16x32_bf16(
          false, afrag[1], false, bfrag, (short)0, acc[1][nt], false, false);
    }
  }

  // Epilogue: tanh for z columns (<256), sigmoid for f columns (>=256).
  // C/D layout: lane l, VGPR r -> (m = r + 8*(l>>4), n = l&15).
#pragma unroll
  for (int p = 0; p < 2; ++p) {
#pragma unroll
    for (int nt = 0; nt < 4; ++nt) {
      const int wnt  = wv * 4 + nt;
      const int ng   = wnt * 16 + ln16;
      const bool isz = (wnt < 16);               // uniform per wave iteration
#pragma unroll
      for (int r = 0; r < 8; ++r) {
        const int m = p * 16 + r + 8 * half;
        float v = acc[p][nt][r];
        v = isz ? tanhf(v) : (1.0f / (1.0f + __expf(-v)));
        zf[(long long)(mt * MROWS + m) * N2 + ng] = v;
      }
    }
  }
}

// --------------------- pass 1: per-chunk (prod f, local h) ------------------
__global__ __launch_bounds__(256)
void scan_chunks(const float* __restrict__ zf, float* __restrict__ aArr,
                 float* __restrict__ bArr) {
  using namespace qrnn;
  const int b = blockIdx.x / CHUNKS;
  const int c = blockIdx.x % CHUNKS;
  const int h = threadIdx.x;
  const float* base = zf + (long long)(b * S + c * CL) * N2;
  float a = 1.0f, acc = 0.0f;
  for (int i = 0; i < CL; ++i) {
    const float z = base[i * N2 + h];
    const float f = base[i * N2 + H + h];
    acc = f * acc + (1.0f - f) * z;
    a *= f;
  }
  const int ch = b * H + h;
  aArr[c * (B * H) + ch] = a;
  bArr[c * (B * H) + ch] = acc;
}

// ------------- pass 2: serial combine over 64 chunks per channel ------------
__global__ void scan_combine(const float* __restrict__ aArr,
                             const float* __restrict__ bArr,
                             float* __restrict__ hin) {
  using namespace qrnn;
  const int ch = blockIdx.x * blockDim.x + threadIdx.x;   // 0..8191
  float hc = 0.0f;                                        // h0 = 0
  for (int c = 0; c < CHUNKS; ++c) {
    hin[c * (B * H) + ch] = hc;
    hc = aArr[c * (B * H) + ch] * hc + bArr[c * (B * H) + ch];
  }
}

// --- pass 3: replay with correct h_in, write hidden, fuse attention logits --
// Wave partials go to LDS inside the loop (no barrier); one barrier at the
// end, then 64 threads finalize the 64 chunk logits.
__global__ __launch_bounds__(256)
void scan_final(const float* __restrict__ zf, const float* __restrict__ hin,
                const float* __restrict__ Mu, float* __restrict__ hidden,
                float* __restrict__ logits) {
  using namespace qrnn;
  __shared__ float sred[CL][8];
  const int b = blockIdx.x / CHUNKS;
  const int c = blockIdx.x % CHUNKS;
  const int h = threadIdx.x;
  const int lane = threadIdx.x & 31;
  const int wv   = threadIdx.x >> 5;
  float hprev = hin[c * (B * H) + b * H + h];
  const float mu = Mu[h];
  const float* base  = zf + (long long)(b * S + c * CL) * N2;
  float*       hbase = hidden + (long long)(b * S + c * CL) * H;
  for (int i = 0; i < CL; ++i) {
    const float z = base[i * N2 + h];
    const float f = base[i * N2 + H + h];
    hprev = f * hprev + (1.0f - f) * z;
    hbase[i * H + h] = hprev;
    const float p = waveReduceSum(hprev * mu);
    if (lane == 0) sred[i][wv] = p;
  }
  __syncthreads();
  if (threadIdx.x < CL) {
    float t = 0.0f;
#pragma unroll
    for (int w = 0; w < 8; ++w) t += sred[threadIdx.x][w];
    logits[b * S + c * CL + threadIdx.x] = t;
  }
}

// ----------------------- softmax over the sequence dim ----------------------
__global__ __launch_bounds__(256)
void softmax_seq(const float* __restrict__ logits, float* __restrict__ attn) {
  using namespace qrnn;
  __shared__ float sbuf[8];
  __shared__ float sbc;
  const int b = blockIdx.x;
  const int t = threadIdx.x;
  float lv[S / 256];
  float mx = -3.4e38f;
#pragma unroll
  for (int i = 0; i < S / 256; ++i) {
    lv[i] = logits[b * S + i * 256 + t];
    mx = fmaxf(mx, lv[i]);
  }
#pragma unroll
  for (int off = 16; off > 0; off >>= 1) mx = fmaxf(mx, __shfl_xor(mx, off, 32));
  if ((t & 31) == 0) sbuf[t >> 5] = mx;
  __syncthreads();
  if (t == 0) {
    float m2 = sbuf[0];
    for (int w = 1; w < 8; ++w) m2 = fmaxf(m2, sbuf[w]);
    sbc = m2;
  }
  __syncthreads();
  mx = sbc;
  float se = 0.0f;
#pragma unroll
  for (int i = 0; i < S / 256; ++i) se += __expf(lv[i] - mx);
  se = waveReduceSum(se);
  if ((t & 31) == 0) sbuf[t >> 5] = se;
  __syncthreads();
  if (t == 0) {
    float s2 = 0.0f;
    for (int w = 0; w < 8; ++w) s2 += sbuf[w];
    sbc = s2;
  }
  __syncthreads();
  const float inv = 1.0f / sbc;
#pragma unroll
  for (int i = 0; i < S / 256; ++i)
    attn[b * S + i * 256 + t] = __expf(lv[i] - mx) * inv;
}

// ------------- attention-weighted pooling + output projection ---------------
__global__ __launch_bounds__(256)
void context_out(const float* __restrict__ hidden, const float* __restrict__ attn,
                 const float* __restrict__ Wout, const float* __restrict__ bout,
                 float* __restrict__ out) {
  using namespace qrnn;
  __shared__ float sbuf[8];
  const int b = blockIdx.x;
  const int h = threadIdx.x;
  const float* hb = hidden + (long long)b * S * H;
  const float* ab = attn + b * S;
  float ctx = 0.0f;
  for (int s = 0; s < S; ++s)
    ctx += ab[s] * hb[(long long)s * H + h];
  float p = waveReduceSum(ctx * Wout[h]);
  if ((h & 31) == 0) sbuf[h >> 5] = p;
  __syncthreads();
  if (h == 0) {
    float t = 0.0f;
    for (int w = 0; w < 8; ++w) t += sbuf[w];
    out[b] = t + bout[0];
  }
}

// ---------------------------------------------------------------------------
extern "C" void kernel_launch(void* const* d_in, const int* in_sizes, int n_in,
                              void* d_out, int out_size, void* d_ws, size_t ws_size,
                              hipStream_t stream) {
  using namespace qrnn;
  (void)in_sizes; (void)n_in; (void)out_size; (void)ws_size;

  const int*   X     = (const int*)d_in[0];
  const float* emb   = (const float*)d_in[1];
  const float* convw = (const float*)d_in[2];
  const float* convb = (const float*)d_in[3];
  const float* Mu    = (const float*)d_in[4];
  const float* Wout  = (const float*)d_in[5];
  const float* bout  = (const float*)d_in[6];
  float* out = (float*)d_out;

  // Workspace carving (total ~410 MB): all buffers fully overwritten each call.
  char* ws = (char*)d_ws;
  size_t off = 0;
  auto take = [&](size_t bytes) -> void* {
    void* p = ws + off;
    off = (off + bytes + 255) & ~(size_t)255;
    return p;
  };
  __bf16* wbf  = (__bf16*)take((size_t)CONVW_ROWS * E * sizeof(__bf16));
  float* zf     = (float*)take((size_t)M * N2 * sizeof(float));
  float* hidden = (float*)take((size_t)M * H * sizeof(float));
  float* aArr   = (float*)take((size_t)CHUNKS * B * H * sizeof(float));
  float* bArr   = (float*)take((size_t)CHUNKS * B * H * sizeof(float));
  float* hin    = (float*)take((size_t)CHUNKS * B * H * sizeof(float));
  float* logits = (float*)take((size_t)B * S * sizeof(float));
  float* attn   = (float*)take((size_t)B * S * sizeof(float));

  convw_to_bf16<<<(CONVW_ROWS * E + 255) / 256, 256, 0, stream>>>(convw, wbf, CONVW_ROWS * E);
  gemm_zf<<<MB, 256, 0, stream>>>(X, emb, wbf, convb, zf);
  scan_chunks<<<B * CHUNKS, 256, 0, stream>>>(zf, aArr, bArr);
  scan_combine<<<(B * H) / 256, 256, 0, stream>>>(aArr, bArr, hin);
  scan_final<<<B * CHUNKS, 256, 0, stream>>>(zf, hin, Mu, hidden, logits);
  softmax_seq<<<B, 256, 0, stream>>>(logits, attn);
  context_out<<<B, 256, 0, stream>>>(hidden, attn, Wout, bout, out);
}
